// DecoderRNN_22041772163512
// MI455X (gfx1250) — compile-verified, matrix operand order checked by reading, not run on hardware
//
#include <hip/hip_runtime.h>
#include <hip/hip_bf16.h>

// ---------------------------------------------------------------------------
// Types for CDNA5 WMMA (wave32): V_WMMA_F32_16X16X32_BF16
// ---------------------------------------------------------------------------
typedef __attribute__((ext_vector_type(16))) __bf16        bf16x16;
typedef __attribute__((ext_vector_type(8)))  float         f32x8;
typedef __attribute__((ext_vector_type(8)))  unsigned int  u32x8;

static constexpr int Bsz = 128;
static constexpr int IN  = 2048;
static constexpr int E   = 512;
static constexpr int H   = 512;
static constexpr int V   = 10000;
static constexpr int T   = 25;
static constexpr int NOBJ = 10;
static constexpr int TS  = T - 1;        // 24 timesteps
static constexpr int MROWS = Bsz * TS;   // 3072

__device__ __forceinline__ unsigned short f2bf(float f) {
    unsigned int u = __float_as_uint(f);
    return (unsigned short)((u + 0x7FFFu + ((u >> 16) & 1u)) >> 16);
}

// Load one 16x32 bf16 WMMA fragment (A or B) from a row-major [rows x K]
// bf16 matrix. lane&15 selects the row (M for A, N for B); K pairs sit at
// element offsets {0,2,4,6,16,18,20,22} + 8*(lane>=16) within the K-block.
// Compiles to two global_load_b128 per fragment.
__device__ __forceinline__ bf16x16 load_frag(const unsigned short* mat, int K,
                                             int row, int k0, int half) {
    const unsigned int* p =
        (const unsigned int*)(mat + (size_t)row * K + k0 + half * 8);
    u32x8 r;
    r[0] = p[0]; r[1] = p[1]; r[2] = p[2];  r[3] = p[3];
    r[4] = p[8]; r[5] = p[9]; r[6] = p[10]; r[7] = p[11];
    return __builtin_bit_cast(bf16x16, r);
}

__device__ __forceinline__ f32x8 wmma_bf16(bf16x16 a, bf16x16 b, f32x8 c) {
    return __builtin_amdgcn_wmma_f32_16x16x32_bf16(
        false, a, false, b, (short)0, c, false, false);
}

__device__ __forceinline__ float sigf(float x) {
    return 1.0f / (1.0f + __expf(-x));
}

// ---------------------------------------------------------------------------
// f32 -> bf16 raw conversion (grid-stride)
// ---------------------------------------------------------------------------
__global__ void k_f2bf(const float* __restrict__ src,
                       unsigned short* __restrict__ dst, int n) {
    for (int i = blockIdx.x * blockDim.x + threadIdx.x; i < n;
         i += gridDim.x * blockDim.x)
        dst[i] = f2bf(src[i]);
}

// ---------------------------------------------------------------------------
// obj_mean into comb_f32[:, 512:1024]
// ---------------------------------------------------------------------------
__global__ void k_objmean(const float* __restrict__ obj,
                          float* __restrict__ comb) {
    int i = blockIdx.x * blockDim.x + threadIdx.x;   // 128*512 threads
    if (i >= Bsz * E) return;
    int b = i >> 9, e = i & 511;
    float s = 0.f;
    const float* p = obj + (size_t)b * NOBJ * E + e;
#pragma unroll
    for (int o = 0; o < NOBJ; ++o) s += p[o * E];
    comb[(size_t)b * 1024 + 512 + e] = s * (1.0f / NOBJ);
}

// ---------------------------------------------------------------------------
// Assemble x_bf16[b][t][0:512]=emb(captions[b][t]), [512:1536]=comb_bf[b]
// ---------------------------------------------------------------------------
__global__ void k_xasm(const int* __restrict__ captions,
                       const float* __restrict__ emb_table,
                       const unsigned short* __restrict__ comb_bf,
                       unsigned short* __restrict__ x) {
    int i = blockIdx.x * blockDim.x + threadIdx.x;   // 128*24*1536
    if (i >= Bsz * TS * 1536) return;
    int b = i / (TS * 1536);
    int r = i - b * (TS * 1536);
    int t = r / 1536;
    int e = r - t * 1536;
    unsigned short v;
    if (e < 512) {
        int tok = captions[b * T + t];
        v = f2bf(emb_table[(size_t)tok * E + e]);
    } else {
        v = comb_bf[(size_t)b * 1024 + (e - 512)];
    }
    x[i] = v;
}

// ---------------------------------------------------------------------------
// Generic WMMA GEMM:  C[m][n] = sum_k A[m][k]*W[n][k] (+bias0[n]+bias1[n])
// A: [M x K] bf16 row-major, W: [N x K] bf16 row-major.
// One wave per 16 x (16*NSUB) tile; NSUB B-subtiles share one A fragment.
// NSUB is a compile-time constant so the k-loop is branch-free and the
// accumulators stay pinned in registers.
// MODE 0: Cf[m*ldc+n] (f32)   MODE 1: Cbf[m*ldc+n] (bf16)
// MODE 2: Cf[(b*24+t)*V + n] with m = t*128+b  (final logits remap)
// ---------------------------------------------------------------------------
template <int NSUB, int MODE>
__global__ void k_gemm(const unsigned short* __restrict__ A,
                       const unsigned short* __restrict__ W,
                       const float* __restrict__ bias0,
                       const float* __restrict__ bias1,
                       float* __restrict__ Cf,
                       unsigned short* __restrict__ Cbf,
                       int K, int ldc, int noff) {
    const int lane = threadIdx.x;          // 0..31
    const int half = lane >> 4;
    const int idx  = lane & 15;
    const int mbase = blockIdx.y * 16;
    const int nblk  = blockIdx.x * (16 * NSUB) + noff;

    f32x8 acc[NSUB];
#pragma unroll
    for (int s = 0; s < NSUB; ++s)
#pragma unroll
        for (int r = 0; r < 8; ++r) acc[s][r] = 0.0f;

    const int arow = mbase + idx;
    for (int k0 = 0; k0 < K; k0 += 32) {
        bf16x16 a = load_frag(A, K, arow, k0, half);
        if (k0 + 32 < K)
            __builtin_prefetch(A + (size_t)arow * K + k0 + 32, 0, 1);
#pragma unroll
        for (int s = 0; s < NSUB; ++s) {
            bf16x16 bfr = load_frag(W, K, nblk + s * 16 + idx, k0, half);
            acc[s] = wmma_bf16(a, bfr, acc[s]);
        }
    }

#pragma unroll
    for (int s = 0; s < NSUB; ++s) {
        int n = nblk + s * 16 + idx;
        float bv = (bias0 ? bias0[n] : 0.0f) + (bias1 ? bias1[n] : 0.0f);
#pragma unroll
        for (int r = 0; r < 8; ++r) {
            int m = mbase + r + half * 8;
            float v = acc[s][r] + bv;
            if (MODE == 0) {
                Cf[(size_t)m * ldc + n] = v;
            } else if (MODE == 1) {
                Cbf[(size_t)m * ldc + n] = f2bf(v);
            } else {
                int b = m & 127, t = m >> 7;           // m = t*128 + b
                Cf[((size_t)b * TS + t) * V + n] = v;
            }
        }
    }
}

// ---------------------------------------------------------------------------
// Fused LSTM step t: gates = gx[:,t,:] + h_in @ W_hh^T ; LSTM cell; write
// c (f32, in-place) and h (bf16) into hs[t]. One wave per 16-batch x
// 16-hidden tile; wave owns all four gate columns (n = gate*512 + j).
// ---------------------------------------------------------------------------
__global__ void k_lstm_step(const float* __restrict__ gx,
                            const unsigned short* __restrict__ h_in,
                            const unsigned short* __restrict__ Whh,
                            float* __restrict__ c,
                            unsigned short* __restrict__ h_out,
                            int t) {
    const int lane = threadIdx.x;
    const int half = lane >> 4;
    const int idx  = lane & 15;
    const int mbase = blockIdx.y * 16;     // batch tile (8 tiles)
    const int jbase = blockIdx.x * 16;     // hidden tile (32 tiles)
    const int j = jbase + idx;

    f32x8 acc[4];
#pragma unroll
    for (int g = 0; g < 4; ++g)
#pragma unroll
        for (int r = 0; r < 8; ++r) {
            int b = mbase + r + half * 8;
            acc[g][r] = gx[((size_t)b * TS + t) * (4 * H) + g * H + j];
        }

    const int arow = mbase + idx;
    for (int k0 = 0; k0 < H; k0 += 32) {
        bf16x16 a = load_frag(h_in, H, arow, k0, half);
#pragma unroll
        for (int g = 0; g < 4; ++g) {
            bf16x16 bfr = load_frag(Whh, H, g * H + jbase + idx, k0, half);
            acc[g] = wmma_bf16(a, bfr, acc[g]);
        }
    }

#pragma unroll
    for (int r = 0; r < 8; ++r) {
        int b = mbase + r + half * 8;
        float iv = sigf(acc[0][r]);
        float fv = sigf(acc[1][r]);
        float gv = tanhf(acc[2][r]);
        float ov = sigf(acc[3][r]);
        size_t ci = (size_t)b * H + j;
        float cn = fv * c[ci] + iv * gv;
        c[ci] = cn;
        h_out[ci] = f2bf(ov * tanhf(cn));
    }
}

// ---------------------------------------------------------------------------
// Host-side orchestration
// ---------------------------------------------------------------------------
extern "C" void kernel_launch(void* const* d_in, const int* in_sizes, int n_in,
                              void* d_out, int out_size, void* d_ws, size_t ws_size,
                              hipStream_t stream) {
    const float* gf      = (const float*)d_in[0];
    const float* objf    = (const float*)d_in[1];
    const int*   caps    = (const int*)  d_in[2];
    const float* emb     = (const float*)d_in[3];
    const float* projW   = (const float*)d_in[4];
    const float* projb   = (const float*)d_in[5];
    const float* ihW     = (const float*)d_in[6];
    const float* ihb     = (const float*)d_in[7];
    const float* icW     = (const float*)d_in[8];
    const float* icb     = (const float*)d_in[9];
    const float* Wih     = (const float*)d_in[10];
    const float* Whh     = (const float*)d_in[11];
    const float* bih     = (const float*)d_in[12];
    const float* bhh     = (const float*)d_in[13];
    const float* fcW     = (const float*)d_in[14];
    const float* fcb     = (const float*)d_in[15];
    float* out = (float*)d_out;

    // Workspace carve-up
    size_t off = 0;
    auto alloc = [&](size_t bytes) -> void* {
        void* p = (char*)d_ws + off;
        off += (bytes + 255) & ~(size_t)255;
        return p;
    };
    unsigned short* projW_bf = (unsigned short*)alloc((size_t)E * IN * 2);
    unsigned short* ihW_bf   = (unsigned short*)alloc((size_t)H * 2 * E * 2);
    unsigned short* icW_bf   = (unsigned short*)alloc((size_t)H * 2 * E * 2);
    unsigned short* Wih_bf   = (unsigned short*)alloc((size_t)4 * H * 3 * E * 2);
    unsigned short* Whh_bf   = (unsigned short*)alloc((size_t)4 * H * H * 2);
    unsigned short* fcW_bf   = (unsigned short*)alloc((size_t)V * H * 2);
    unsigned short* gf_bf    = (unsigned short*)alloc((size_t)Bsz * IN * 2);
    float*          comb_f   = (float*)        alloc((size_t)Bsz * 1024 * 4);
    unsigned short* comb_bf  = (unsigned short*)alloc((size_t)Bsz * 1024 * 2);
    unsigned short* x_bf     = (unsigned short*)alloc((size_t)Bsz * TS * 1536 * 2);
    float*          gx       = (float*)        alloc((size_t)Bsz * TS * 2048 * 4);
    unsigned short* h0_bf    = (unsigned short*)alloc((size_t)Bsz * H * 2);
    float*          c_f      = (float*)        alloc((size_t)Bsz * H * 4);
    unsigned short* hs_bf    = (unsigned short*)alloc((size_t)TS * Bsz * H * 2);

    auto cvt = [&](const float* s, unsigned short* d, int n) {
        int blk = 256, grd = (n + blk - 1) / blk;
        if (grd > 4096) grd = 4096;
        k_f2bf<<<grd, blk, 0, stream>>>(s, d, n);
    };

    // 1. Down-convert weights & global features to bf16
    cvt(projW, projW_bf, E * IN);
    cvt(ihW,   ihW_bf,   H * 2 * E);
    cvt(icW,   icW_bf,   H * 2 * E);
    cvt(Wih,   Wih_bf,   4 * H * 3 * E);
    cvt(Whh,   Whh_bf,   4 * H * H);
    cvt(fcW,   fcW_bf,   V * H);
    cvt(gf,    gf_bf,    Bsz * IN);

    // 2. feats = gf @ projW^T + projb  -> comb_f[:, 0:512]
    k_gemm<4, 0><<<dim3(E / 64, Bsz / 16), 32, 0, stream>>>(
        gf_bf, projW_bf, projb, nullptr, comb_f, nullptr, IN, 1024, 0);

    // 3. obj mean -> comb_f[:, 512:1024]
    k_objmean<<<(Bsz * E + 255) / 256, 256, 0, stream>>>(objf, comb_f);

    // 4. comb -> bf16
    cvt(comb_f, comb_bf, Bsz * 1024);

    // 5. h0 (bf16) and c0 (f32)
    k_gemm<4, 1><<<dim3(H / 64, Bsz / 16), 32, 0, stream>>>(
        comb_bf, ihW_bf, ihb, nullptr, nullptr, h0_bf, 2 * E, H, 0);
    k_gemm<4, 0><<<dim3(H / 64, Bsz / 16), 32, 0, stream>>>(
        comb_bf, icW_bf, icb, nullptr, c_f, nullptr, 2 * E, H, 0);

    // 6. assemble x (emb gather + comb broadcast), bf16
    {
        int n = Bsz * TS * 1536;
        k_xasm<<<(n + 255) / 256, 256, 0, stream>>>(caps, emb, comb_bf, x_bf);
    }

    // 7. gx = x @ W_ih^T + b_ih + b_hh   (3072 x 2048, K=1536)
    k_gemm<4, 0><<<dim3(2048 / 64, MROWS / 16), 32, 0, stream>>>(
        x_bf, Wih_bf, bih, bhh, gx, nullptr, 3 * E, 4 * H, 0);

    // 8. 24 serial fused LSTM steps; h for step t written into hs[t]
    for (int t = 0; t < TS; ++t) {
        const unsigned short* h_in =
            (t == 0) ? h0_bf : (hs_bf + (size_t)(t - 1) * Bsz * H);
        unsigned short* h_out = hs_bf + (size_t)t * Bsz * H;
        k_lstm_step<<<dim3(H / 16, Bsz / 16), 32, 0, stream>>>(
            gx, h_in, Whh_bf, c_f, h_out, t);
    }

    // 9. logits: out[b][t][v] = hs[t][b] . fc_W[v] + fc_b[v]
    //    M = 3072 rows (t-major), N = 10000 = 156*64 + 16
    k_gemm<4, 2><<<dim3(156, MROWS / 16), 32, 0, stream>>>(
        hs_bf, fcW_bf, fcb, nullptr, out, nullptr, H, V, 0);
    k_gemm<1, 2><<<dim3(1, MROWS / 16), 32, 0, stream>>>(
        hs_bf, fcW_bf, fcb, nullptr, out, nullptr, H, V, 156 * 64);
}
